// SimpleAttention_1434519077254
// MI455X (gfx1250) — matrix operand order repaired and checked by reading, not compiled, not run on hardware
//
#include <hip/hip_runtime.h>
#include <hip/hip_bf16.h>
#include <math.h>

typedef __attribute__((ext_vector_type(16))) _Float16 v16h;
typedef __attribute__((ext_vector_type(8)))  _Float16 v8h;
typedef __attribute__((ext_vector_type(8)))  float    v8f;
typedef __attribute__((ext_vector_type(4)))  unsigned int v4u;
typedef __attribute__((ext_vector_type(8)))  int      v8i;
typedef __attribute__((ext_vector_type(4)))  int      v4i;

#define B_    8
#define S_    2048
#define H_    256
#define MROWS (B_ * S_)   // 16384

#define KROW  264         // f16 elems per K LDS row: 256 + 8 pad (528B, conflict-free)
#define VROW  40          // f16 elems per V LDS row: 32 + 8 pad  (80B,  conflict-free)

// Build a 16-half A/B-style fragment from two contiguous 8-half (16B) chunks.
__device__ __forceinline__ v16h frag_from_halves(const _Float16* lo, const _Float16* hi) {
  v8h a = *(const v8h*)lo;
  v8h b = *(const v8h*)hi;
  v16h r;
#pragma unroll
  for (int i = 0; i < 8; ++i) { r[i] = a[i]; r[i + 8] = b[i]; }
  return r;
}

__device__ __forceinline__ v8f wmma_f16(v16h a, v16h b, v8f c) {
  return __builtin_amdgcn_wmma_f32_16x16x32_f16(false, a, false, b, (short)0, c,
                                                false, false);
}

// ---------------------------------------------------------------------------
// TDM: async DMA of a 2D f16 tile (global -> LDS) with LDS row padding.
// D# packed per CDNA5 ISA 8.3/8.4: group0 = {flags, lds_addr, global_addr,
// type=2}; group1 = {data_size/pad, tensor dims, tile dims, dim0 stride}.
// Groups 2/3 (and trailing group) zero (<=2D tensor). Tracked by TENSORcnt.
// ---------------------------------------------------------------------------
__device__ __forceinline__ void tdm_load_2d(unsigned lds_addr, const void* gptr,
                                            unsigned tensor_d0, unsigned tensor_d1,
                                            unsigned tile_d0, unsigned tile_d1,
                                            unsigned stride0,
                                            unsigned pad_interval,
                                            unsigned pad_amount) {
  unsigned long long ga = (unsigned long long)(size_t)gptr;
  v4u g0;
  g0[0] = 1u;                                            // count=1 (valid), user
  g0[1] = lds_addr;                                      // LDS byte address
  g0[2] = (unsigned)(ga & 0xffffffffu);                  // global addr [31:0]
  g0[3] = (unsigned)((ga >> 32) & 0x1ffffffu) | (2u << 30);  // [56:32] | type=2
  v8i g1;
  g1[0] = (int)((1u << 16)                               // data_size = 2 bytes
              | (1u << 20)                               // pad_enable
              | (pad_interval << 22) | (pad_amount << 25));
  g1[1] = (int)((tensor_d0 & 0xffffu) << 16);            // tensor_dim0[15:0]
  g1[2] = (int)((tensor_d0 >> 16) | ((tensor_d1 & 0xffffu) << 16));
  g1[3] = (int)((tensor_d1 >> 16) | (tile_d0 << 16));    // tile_dim0
  g1[4] = (int)tile_d1;                                  // tile_dim1 (tile_dim2=0)
  g1[5] = (int)stride0;                                  // tensor_dim0_stride[31:0]
  g1[6] = 0;                                             // stride0 hi, stride1 lo
  g1[7] = 0;
  v4i z4 = {0, 0, 0, 0};
  v8i z8 = {0, 0, 0, 0, 0, 0, 0, 0};
  __builtin_amdgcn_tensor_load_to_lds(g0, g1, z4, z4, z8, 0);
}

// ---------------------------------------------------------------------------
// Kernel 0: W (f32, [H][D]) -> Wt (f16, [D][H]) for all 3 weights.
// ---------------------------------------------------------------------------
__global__ void wtrans_kernel(const float* __restrict__ Wq,
                              const float* __restrict__ Wk,
                              const float* __restrict__ Wv,
                              _Float16* __restrict__ Wtq,
                              _Float16* __restrict__ Wtk,
                              _Float16* __restrict__ Wtv) {
  int id = blockIdx.x * blockDim.x + threadIdx.x;
  int which = id >> 16;
  int r = id & 65535;
  int d = r >> 8;
  int h = r & 255;
  const float* W = (which == 0) ? Wq : (which == 1) ? Wk : Wv;
  _Float16* Wt   = (which == 0) ? Wtq : (which == 1) ? Wtk : Wtv;
  Wt[d * H_ + h] = (_Float16)W[h * H_ + d];
}

// ---------------------------------------------------------------------------
// Kernel 1: projections. z=0: Qh = q@Wq ; z=1: Kh = k@Wk (row-major f16)
//           z=2: Vt = (v@Wv)^T stored [B][H][S] f16.
// ---------------------------------------------------------------------------
__global__ __launch_bounds__(128) void proj_kernel(
    const float* __restrict__ q, const float* __restrict__ k,
    const float* __restrict__ v,
    const _Float16* __restrict__ Wtq, const _Float16* __restrict__ Wtk,
    const _Float16* __restrict__ Wtv,
    _Float16* __restrict__ Qh, _Float16* __restrict__ Kh,
    _Float16* __restrict__ VT) {
  int z = blockIdx.y;
  const float* X     = (z == 0) ? q   : (z == 1) ? k   : v;
  const _Float16* Wt = (z == 0) ? Wtq : (z == 1) ? Wtk : Wtv;

  int lane = threadIdx.x & 31;
  int wave = threadIdx.x >> 5;
  int tile = blockIdx.x * 4 + wave;
  int m0 = (tile >> 4) << 4;
  int n0 = (tile & 15) << 4;
  int ln = lane & 15;
  int hi = lane >> 4;
  int kA = hi << 3;
  int kB = hi << 4;

  const float*    xrow = X  + (size_t)(m0 + ln) * H_;
  const _Float16* wrow = Wt + (size_t)(n0 + ln) * H_;

  v8f c = {};
#pragma unroll
  for (int k0 = 0; k0 < H_; k0 += 32) {
    float4 x0 = *(const float4*)(xrow + k0 + kA);
    float4 x1 = *(const float4*)(xrow + k0 + kA + 4);
    float4 x2 = *(const float4*)(xrow + k0 + kA + 16);
    float4 x3 = *(const float4*)(xrow + k0 + kA + 20);
    v16h a;
    a[0]=(_Float16)x0.x; a[1]=(_Float16)x0.y; a[2]=(_Float16)x0.z; a[3]=(_Float16)x0.w;
    a[4]=(_Float16)x1.x; a[5]=(_Float16)x1.y; a[6]=(_Float16)x1.z; a[7]=(_Float16)x1.w;
    a[8]=(_Float16)x2.x; a[9]=(_Float16)x2.y; a[10]=(_Float16)x2.z; a[11]=(_Float16)x2.w;
    a[12]=(_Float16)x3.x; a[13]=(_Float16)x3.y; a[14]=(_Float16)x3.z; a[15]=(_Float16)x3.w;
    v16h b = frag_from_halves(wrow + k0 + kA, wrow + k0 + kA + 16);
    c = wmma_f16(a, b, c);
  }

  if (z < 2) {
    _Float16* Y = (z == 0) ? Qh : Kh;
#pragma unroll
    for (int r = 0; r < 8; ++r) {
      int m = m0 + r + (hi << 3);
      Y[(size_t)m * H_ + n0 + ln] = (_Float16)c[r];
    }
  } else {
#pragma unroll
    for (int r = 0; r < 8; ++r) {
      int m  = m0 + r + (hi << 3);
      int bb = m >> 11;
      int s  = m & (S_ - 1);
      VT[((size_t)bb * H_ + n0 + ln) * S_ + s] = (_Float16)c[r];
    }
  }
}

// ---------------------------------------------------------------------------
// Kernel 2: flash attention, TDM double-buffered K/V tiles in LDS.
// Block = 4 waves, all on the same batch; wave w owns 16 query rows.
// Per 32-key chunk: 16 WMMA (QK^T) + online softmax + 16 WMMA (PV).
// ---------------------------------------------------------------------------
__global__ __launch_bounds__(128, 1) void attn_kernel(
    const _Float16* __restrict__ Qh, const _Float16* __restrict__ Kh,
    const _Float16* __restrict__ VT, const unsigned char* __restrict__ mask,
    float* __restrict__ out) {
  __shared__ __align__(16) _Float16 Klds[2][32 * KROW];   // 2 x 16.5 KB
  __shared__ __align__(16) _Float16 Vlds[2][256 * VROW];  // 2 x 20 KB
  __shared__ __align__(32) _Float16 psh[4 * 512];         // P staging, per wave

  const float norm = 0.0625f;       // 1/sqrt(256)
  const float NEG  = -3.0e38f;

  int lane  = threadIdx.x & 31;
  int wave  = threadIdx.x >> 5;
  int wid   = blockIdx.x * 4 + wave;
  int batch = wid >> 7;
  int q0    = (wid & 127) << 4;
  int ln = lane & 15;
  int hi = lane >> 4;
  int kA = hi << 3;
  int kB = hi << 4;
  _Float16* pl = psh + wave * 512;

  const _Float16* qrow  = Qh + ((size_t)batch * S_ + q0 + ln) * H_ + kB;
  const unsigned char* mbase = mask + ((size_t)batch * S_ + q0) * (size_t)S_;
  const _Float16* kgbase = Kh + (size_t)batch * S_ * H_;
  const _Float16* vgbase = VT + (size_t)batch * H_ * S_;

  unsigned klds0 = (unsigned)(size_t)(void*)&Klds[0][0];
  unsigned klds1 = (unsigned)(size_t)(void*)&Klds[1][0];
  unsigned vlds0 = (unsigned)(size_t)(void*)&Vlds[0][0];
  unsigned vlds1 = (unsigned)(size_t)(void*)&Vlds[1][0];

  v8f acc[16];
#pragma unroll
  for (int t = 0; t < 16; ++t) acc[t] = (v8f){0, 0, 0, 0, 0, 0, 0, 0};
  float mrun[8], lrun[8];
#pragma unroll
  for (int r = 0; r < 8; ++r) { mrun[r] = NEG; lrun[r] = 0.f; }

  // Prologue: DMA chunk 0 into buffer 0 (K: 32x256 rows; V: 256x32 rows).
  if (wave == 0) {
    tdm_load_2d(klds0, kgbase, H_, 32, H_, 32, H_, /*pad:128dw*/6, /*4dw*/3);
    tdm_load_2d(vlds0, vgbase, 32, H_, 32, H_, S_, /*pad:16dw*/3, /*4dw*/3);
  }

  for (int key0 = 0; key0 < S_; key0 += 32) {
    int cur = (key0 >> 5) & 1;
    // Current chunk's DMA complete (TENSORcnt in-order, all issued for cur).
    if (wave == 0) __builtin_amdgcn_s_wait_tensorcnt((short)0);
    __syncthreads();   // data ready; previous readers of the other buffer done
    // Kick off next chunk into the other buffer; overlaps with compute below.
    if (wave == 0 && key0 + 32 < S_) {
      int nk = key0 + 32;
      tdm_load_2d(cur ? klds0 : klds1, kgbase + (size_t)nk * H_,
                  H_, 32, H_, 32, H_, 6, 3);
      tdm_load_2d(cur ? vlds0 : vlds1, vgbase + nk,
                  32, H_, 32, H_, S_, 3, 3);
    }
    const _Float16* Kl = &Klds[cur][0];
    const _Float16* Vl = &Vlds[cur][0];

    // ---- S tile (16 q x 32 k) = Q @ K^T, B-frags from LDS ----
    v8f s0 = {}, s1 = {};
    const _Float16* kr0 = Kl + (size_t)ln * KROW + kB;         // keys 0..15
    const _Float16* kr1 = Kl + (size_t)(16 + ln) * KROW + kB;  // keys 16..31
#pragma unroll
    for (int kc = 0; kc < H_; kc += 32) {
      v16h a  = frag_from_halves(qrow + kc, qrow + kc + 8);
      v16h b0 = frag_from_halves(kr0 + kc, kr0 + kc + 8);
      v16h b1 = frag_from_halves(kr1 + kc, kr1 + kc + 8);
      s0 = wmma_f16(a, b0, s0);
      s1 = wmma_f16(a, b1, s1);
    }

    // ---- mask + online softmax (row stats across 16-lane halves) ----
    float alpha[8];
#pragma unroll
    for (int r = 0; r < 8; ++r) {
      const unsigned char* mrow = mbase + (size_t)(r + (hi << 3)) * S_ + key0;
      float v0 = mrow[ln]      ? NEG : s0[r] * norm;
      float v1 = mrow[16 + ln] ? NEG : s1[r] * norm;
      float mx = fmaxf(v0, v1);
#pragma unroll
      for (int j = 1; j < 16; j <<= 1) mx = fmaxf(mx, __shfl_xor(mx, j, 32));
      float mnew = fmaxf(mrun[r], mx);
      alpha[r] = __expf(mrun[r] - mnew);
      mrun[r]  = mnew;
      float p0 = (v0 <= -1.0e37f) ? 0.f : __expf(v0 - mnew);
      float p1 = (v1 <= -1.0e37f) ? 0.f : __expf(v1 - mnew);
      float rs = p0 + p1;
#pragma unroll
      for (int j = 1; j < 16; j <<= 1) rs += __shfl_xor(rs, j, 32);
      lrun[r] = lrun[r] * alpha[r] + rs;
      s0[r] = p0; s1[r] = p1;
    }

    // ---- rescale running accumulators ----
#pragma unroll
    for (int t = 0; t < 16; ++t)
#pragma unroll
      for (int r = 0; r < 8; ++r) acc[t][r] *= alpha[r];

    // ---- P: C-layout f32 -> LDS -> A-layout f16 ----
#pragma unroll
    for (int r = 0; r < 8; ++r) {
      pl[(r + (hi << 3)) * 32 + ln]      = (_Float16)s0[r];
      pl[(r + (hi << 3)) * 32 + 16 + ln] = (_Float16)s1[r];
    }
    asm volatile("s_wait_dscnt 0" ::: "memory");
    const _Float16* pr = pl + ln * 32 + kB;
    v16h pa = frag_from_halves(pr, pr + 8);
    asm volatile("" ::: "memory");

    // ---- acc += P @ V, B-frags from LDS (pre-transposed V rows) ----
#pragma unroll
    for (int t = 0; t < 16; ++t) {
      const _Float16* vr = Vl + (size_t)(t * 16 + ln) * VROW + kB;
      v16h vb = frag_from_halves(vr, vr + 8);
      acc[t] = wmma_f16(pa, vb, acc[t]);
    }
  }

  // ---- epilogue: 1/l scaling; fully-masked rows -> 0 (nan_to_num) ----
  float scale[8];
#pragma unroll
  for (int r = 0; r < 8; ++r)
    scale[r] = (lrun[r] > 0.f && mrun[r] > -1.0e37f) ? 1.f / lrun[r] : 0.f;
#pragma unroll
  for (int t = 0; t < 16; ++t)
#pragma unroll
    for (int r = 0; r < 8; ++r)
      out[((size_t)batch * S_ + q0 + r + (hi << 3)) * H_ + t * 16 + ln] =
          acc[t][r] * scale[r];
}

// ---------------------------------------------------------------------------
extern "C" void kernel_launch(void* const* d_in, const int* in_sizes, int n_in,
                              void* d_out, int out_size, void* d_ws,
                              size_t ws_size, hipStream_t stream) {
  (void)in_sizes; (void)n_in; (void)out_size; (void)ws_size;
  // setup_inputs order: k, q, v, mask, W_query, W_key, W_val
  const float* k_in = (const float*)d_in[0];
  const float* q_in = (const float*)d_in[1];
  const float* v_in = (const float*)d_in[2];
  const unsigned char* mask = (const unsigned char*)d_in[3];
  const float* Wq = (const float*)d_in[4];
  const float* Wk = (const float*)d_in[5];
  const float* Wv = (const float*)d_in[6];
  float* out = (float*)d_out;

  _Float16* Qh  = (_Float16*)d_ws;                     // 8 MB
  _Float16* Kh  = Qh  + (size_t)MROWS * H_;            // 8 MB
  _Float16* VT  = Kh  + (size_t)MROWS * H_;            // 8 MB
  _Float16* Wtq = VT  + (size_t)MROWS * H_;            // 128 KB x3
  _Float16* Wtk = Wtq + H_ * H_;
  _Float16* Wtv = Wtk + H_ * H_;

  wtrans_kernel<<<dim3(3 * H_ * H_ / 256), 256, 0, stream>>>(Wq, Wk, Wv, Wtq,
                                                             Wtk, Wtv);
  proj_kernel<<<dim3((MROWS / 16) * (H_ / 16) / 4, 3), 128, 0, stream>>>(
      q_in, k_in, v_in, Wtq, Wtk, Wtv, Qh, Kh, VT);
  attn_kernel<<<dim3(MROWS / 16 / 4), 128, 0, stream>>>(Qh, Kh, VT, mask, out);
}
